// DMCNN_VD_41738492182827
// MI455X (gfx1250) — matrix operand order repaired and checked by reading, not compiled, not run on hardware
//
#include <hip/hip_runtime.h>

typedef __attribute__((ext_vector_type(16))) _Float16 v16h;
typedef __attribute__((ext_vector_type(8)))  float    v8f;

#define B_   4
#define H_   256
#define W_   256
#define NCC_ 64
#define EPS_ 1e-5f
#define NHID 18

#define WFRAG_HALFS (9 * 2 * 4 * 32 * 16)          // 36864 halfs = 73728 B
#define WFRAG_BYTES (WFRAG_HALFS * 2)
#define STAGE_BYTES (8 * 32 * NCC_ * 2)            // 8 waves * 4KB = 32768 B
#define SMEM_BYTES  (WFRAG_BYTES + STAGE_BYTES)    // 106496 B dynamic LDS

__device__ __forceinline__ float selu_f(float x) {
    const float l = 1.0507009873554805f;
    const float a = 1.6732632423543772f;
    return x > 0.f ? l * x : l * a * (__expf(x) - 1.f);
}

// ---------------------------------------------------------------------------
// Pack hidden-layer weights (BN folded) into per-lane WMMA B-fragment layout:
// frag index = ((tap*2 + ks)*4 + nt)*32 + lane ; 16 halfs per entry.
// B (32x16, f16): lane -> column n = lane&15, K-half = lane>>4;
//                 within lane, t=0..15 -> K = (lane>>4)*16 + t (this code's k order, as for A).
// Also emits folded bias (64 floats).
// ---------------------------------------------------------------------------
__global__ void pack_hidden_weights(const float* __restrict__ w_h,
                                    const float* __restrict__ b_h,
                                    const float* __restrict__ g_h,
                                    const float* __restrict__ be_h,
                                    const float* __restrict__ m_h,
                                    const float* __restrict__ v_h,
                                    _Float16* __restrict__ wfrag,
                                    float* __restrict__ bias) {
    const int idx = blockIdx.x * blockDim.x + threadIdx.x;
    if (idx < NCC_) {
        float sc = g_h[idx] * rsqrtf(v_h[idx] + EPS_);
        bias[idx] = (b_h[idx] - m_h[idx]) * sc + be_h[idx];
    }
    if (idx >= 9 * 2 * 4 * 32) return;
    const int lane = idx & 31;
    const int nt   = (idx >> 5) & 3;
    const int ks   = (idx >> 7) & 1;
    const int tap  = idx >> 8;              // 0..8
    const int n  = lane & 15;
    const int hi = lane >> 4;
    const int oc = nt * 16 + n;
    const int ty = tap / 3, tx = tap % 3;
    const float sc = g_h[oc] * rsqrtf(v_h[oc] + EPS_);
    _Float16* dst = wfrag + (size_t)idx * 16;
#pragma unroll
    for (int t = 0; t < 16; ++t) {
        const int c = ks * 32 + hi * 16 + t;            // input channel 0..63
        dst[t] = (_Float16)(w_h[(oc * NCC_ + c) * 9 + ty * 3 + tx] * sc);
    }
}

// ---------------------------------------------------------------------------
// Input conv 3->64 + BN + SELU, writes NHWC f16 activations. Tiny FLOP count;
// plain VALU kernel, one thread per (pixel, oc).
// ---------------------------------------------------------------------------
__global__ void input_conv_kernel(const float* __restrict__ X,
                                  const float* __restrict__ w_in,
                                  const float* __restrict__ b_in,
                                  const float* __restrict__ g,
                                  const float* __restrict__ be,
                                  const float* __restrict__ m,
                                  const float* __restrict__ v,
                                  _Float16* __restrict__ act) {
    const int idx = blockIdx.x * blockDim.x + threadIdx.x;   // B*H*W*64
    const int oc = idx & 63;
    const int p  = idx >> 6;
    const int x  = p & (W_ - 1);
    const int y  = (p >> 8) & (H_ - 1);
    const int b  = p >> 16;
    float s = 0.f;
#pragma unroll
    for (int c = 0; c < 3; ++c)
#pragma unroll
        for (int ty = 0; ty < 3; ++ty) {
            const int yy = y + ty - 1;
            if ((unsigned)yy >= H_) continue;
#pragma unroll
            for (int tx = 0; tx < 3; ++tx) {
                const int xx = x + tx - 1;
                if ((unsigned)xx >= W_) continue;
                s += X[((b * 3 + c) * H_ + yy) * W_ + xx] *
                     w_in[(oc * 3 + c) * 9 + ty * 3 + tx];
            }
        }
    const float sc  = g[oc] * rsqrtf(v[oc] + EPS_);
    const float val = selu_f((s + b_in[oc] - m[oc]) * sc + be[oc]);
    act[(size_t)p * NCC_ + oc] = (_Float16)val;
}

// ---------------------------------------------------------------------------
// Hidden layer: implicit-GEMM 3x3 conv 64->64 via v_wmma_f32_16x16x32_f16.
// One wave = 32 pixels (along W) x 64 output channels:
//   2 M-tiles x 4 N-tiles x (9 taps * 2 ksteps) = 144 WMMA / wave.
// Each LDS B fragment feeds two WMMAs (M-tile reuse). Weights staged in LDS;
// output staged through LDS: lane L owns pixel L -> 8 coalesced b128 stores.
// ---------------------------------------------------------------------------
__global__ void __launch_bounds__(256)
hidden_conv_wmma(const _Float16* __restrict__ actIn,
                 _Float16* __restrict__ actOut,
                 const _Float16* __restrict__ wfrag,
                 const float* __restrict__ bias) {
    extern __shared__ char smem_raw[];
    _Float16* wlds  = (_Float16*)smem_raw;                 // 73728 B weights
    _Float16* stage = (_Float16*)(smem_raw + WFRAG_BYTES); // 32 KB staging

    const int tid    = threadIdx.x;
    const int lane   = tid & 31;
    const int wave   = tid >> 5;
    const int waveId = (blockIdx.x * blockDim.x + tid) >> 5; // 0..8191
    const int b  = waveId >> 11;          // 2048 tiles per image
    const int h  = (waveId >> 3) & 255;
    const int w0 = (waveId & 7) << 5;     // 32-pixel tile along W
    const int n  = lane & 15;
    const int hi = lane >> 4;

    // Cooperative load of all B fragments into LDS (once per block).
    {
        const uint4* s4 = (const uint4*)wfrag;
        uint4*       d4 = (uint4*)wlds;
#pragma unroll
        for (int i = 0; i < WFRAG_BYTES / 16 / 256; ++i)   // 18 iters
            d4[tid + i * 256] = s4[tid + i * 256];
    }
    __syncthreads();

    v8f acc[2][4];
#pragma unroll
    for (int mt = 0; mt < 2; ++mt)
#pragma unroll
        for (int nt = 0; nt < 4; ++nt) acc[mt][nt] = v8f{};

    const int xm0 = w0 + n;        // A row m = lane&15 of M-tile 0
    const int xm1 = w0 + 16 + n;   // A row m of M-tile 1

    // Prefetch the three activation rows this tile touches.
#pragma unroll
    for (int ty = 0; ty < 3; ++ty) {
        const int yy = h + ty - 1;
        if ((unsigned)yy < H_)
            __builtin_prefetch(actIn + (size_t)((b * H_ + yy) * W_ + xm0) * NCC_,
                               0, 1);
    }

    union AF { v16h v; uint4 q[2]; };

    for (int tap = 0; tap < 9; ++tap) {
        const int ty = tap / 3, tx = tap % 3;
        const int yy = h + ty - 1;
        const bool vy  = (unsigned)yy < H_;
        const int xx0 = xm0 + tx - 1;
        const int xx1 = xm1 + tx - 1;
        const bool ok0 = vy && ((unsigned)xx0 < W_);
        const bool ok1 = vy && ((unsigned)xx1 < W_);
        const _Float16* row0 = actIn + (size_t)((b * H_ + yy) * W_ + xx0) * NCC_;
        const _Float16* row1 = actIn + (size_t)((b * H_ + yy) * W_ + xx1) * NCC_;
#pragma unroll
        for (int ks = 0; ks < 2; ++ks) {
            // A k order used here (same as B pack): lane hi=0 -> K 0-15,
            //                     lane hi=1 -> K 16-31.
            AF a0, a1;
            if (ok0) {
                const uint4* p4 = (const uint4*)(row0 + ks * 32 + hi * 16);
                a0.q[0] = p4[0];
                a0.q[1] = p4[1];
            } else {
                a0.q[0] = make_uint4(0u, 0u, 0u, 0u);
                a0.q[1] = make_uint4(0u, 0u, 0u, 0u);
            }
            if (ok1) {
                const uint4* p4 = (const uint4*)(row1 + ks * 32 + hi * 16);
                a1.q[0] = p4[0];
                a1.q[1] = p4[1];
            } else {
                a1.q[0] = make_uint4(0u, 0u, 0u, 0u);
                a1.q[1] = make_uint4(0u, 0u, 0u, 0u);
            }
            const _Float16* wp =
                wlds + (size_t)(((tap * 2 + ks) * 4) * 32 + lane) * 16;
#pragma unroll
            for (int nt = 0; nt < 4; ++nt) {
                v16h bf = *(const v16h*)(wp + (size_t)nt * 32 * 16);
                acc[0][nt] = __builtin_amdgcn_wmma_f32_16x16x32_f16(
                    false, a0.v, false, bf, (short)0, acc[0][nt], false, false);
                acc[1][nt] = __builtin_amdgcn_wmma_f32_16x16x32_f16(
                    false, a1.v, false, bf, (short)0, acc[1][nt], false, false);
            }
        }
    }

    // Epilogue: bias + SELU -> LDS staging (C/D layout: VGPR vv, lane ->
    // pixel m = mt*16 + vv + 8*hi, channel = nt*16 + n).
    _Float16* st = stage + wave * (32 * NCC_);
#pragma unroll
    for (int mt = 0; mt < 2; ++mt)
#pragma unroll
        for (int nt = 0; nt < 4; ++nt) {
            const int oc  = nt * 16 + n;
            const float bi = bias[oc];
#pragma unroll
            for (int vv = 0; vv < 8; ++vv) {
                const int mm = mt * 16 + vv + hi * 8;
                st[mm * NCC_ + oc] = (_Float16)selu_f(acc[mt][nt][vv] + bi);
            }
        }
    __syncthreads();

    // Writeback: lane L owns pixel L (128 B contiguous) -> 8 b128 stores,
    // 4 KB contiguous per wave.
    const uint4* s4 = (const uint4*)(st + lane * NCC_);
    uint4* o4 = (uint4*)(actOut + (size_t)((b * H_ + h) * W_ + w0 + lane) * NCC_);
#pragma unroll
    for (int i = 0; i < 8; ++i) o4[i] = s4[i];
}

// ---------------------------------------------------------------------------
// Output conv 64->3 + BN + SELU + zero-interpolation residual, f32 NCHW out.
// ---------------------------------------------------------------------------
__global__ void output_conv_kernel(const _Float16* __restrict__ act,
                                   const float* __restrict__ X,
                                   const float* __restrict__ w_out,
                                   const float* __restrict__ b_out,
                                   const float* __restrict__ g,
                                   const float* __restrict__ be,
                                   const float* __restrict__ m,
                                   const float* __restrict__ v,
                                   float* __restrict__ out) {
    const int p = blockIdx.x * blockDim.x + threadIdx.x;  // B*H*W
    const int x = p & (W_ - 1);
    const int y = (p >> 8) & (H_ - 1);
    const int b = p >> 16;

    float a0 = 0.f, a1 = 0.f, a2 = 0.f;
    for (int ty = 0; ty < 3; ++ty) {
        const int yy = y + ty - 1;
        if ((unsigned)yy >= H_) continue;
        for (int tx = 0; tx < 3; ++tx) {
            const int xx = x + tx - 1;
            if ((unsigned)xx >= W_) continue;
            const _Float16* ap = act + (size_t)((b * H_ + yy) * W_ + xx) * NCC_;
            const int wb = ty * 3 + tx;
#pragma unroll 8
            for (int c = 0; c < NCC_; ++c) {
                const float av = (float)ap[c];
                a0 += av * w_out[(0 * NCC_ + c) * 9 + wb];
                a1 += av * w_out[(1 * NCC_ + c) * 9 + wb];
                a2 += av * w_out[(2 * NCC_ + c) * 9 + wb];
            }
        }
    }
    float r[3] = {a0, a1, a2};
#pragma unroll
    for (int oc = 0; oc < 3; ++oc) {
        const float sc  = g[oc] * rsqrtf(v[oc] + EPS_);
        const float val = selu_f((r[oc] + b_out[oc] - m[oc]) * sc + be[oc]);
        // residual: interpolate_zeros(X)
        const float* Xc = X + (size_t)(b * 3 + oc) * H_ * W_;
        const float xc  = Xc[y * W_ + x];
        float res;
        if (xc != 0.f) {
            res = xc;
        } else {
            const float lft = (x > 0)      ? Xc[y * W_ + x - 1]   : 0.f;
            const float rgt = (x < W_ - 1) ? Xc[y * W_ + x + 1]   : 0.f;
            const float top = (y > 0)      ? Xc[(y - 1) * W_ + x] : 0.f;
            const float bot = (y < H_ - 1) ? Xc[(y + 1) * W_ + x] : 0.f;
            const float nsum = lft + rgt + top + bot;
            const float ncnt = (lft > 0.f ? 1.f : 0.f) + (rgt > 0.f ? 1.f : 0.f) +
                               (top > 0.f ? 1.f : 0.f) + (bot > 0.f ? 1.f : 0.f);
            res = (ncnt > 0.f) ? nsum / fmaxf(ncnt, 1.f) : 0.f;
        }
        out[((size_t)(b * 3 + oc) * H_ + y) * W_ + x] = val + res;
    }
}

// ---------------------------------------------------------------------------
extern "C" void kernel_launch(void* const* d_in, const int* in_sizes, int n_in,
                              void* d_out, int out_size, void* d_ws, size_t ws_size,
                              hipStream_t stream) {
    const float* X      = (const float*)d_in[0];
    const float* w_in   = (const float*)d_in[1];
    const float* b_in   = (const float*)d_in[2];
    const float* g_in   = (const float*)d_in[3];
    const float* be_in  = (const float*)d_in[4];
    const float* m_in   = (const float*)d_in[5];
    const float* v_in   = (const float*)d_in[6];
    const float* w_h    = (const float*)d_in[7];
    const float* b_h    = (const float*)d_in[8];
    const float* g_h    = (const float*)d_in[9];
    const float* be_h   = (const float*)d_in[10];
    const float* m_h    = (const float*)d_in[11];
    const float* v_h    = (const float*)d_in[12];
    const float* w_out  = (const float*)d_in[13];
    const float* b_out  = (const float*)d_in[14];
    const float* g_out  = (const float*)d_in[15];
    const float* be_out = (const float*)d_in[16];
    const float* m_out  = (const float*)d_in[17];
    const float* v_out  = (const float*)d_in[18];
    float* out = (float*)d_out;

    char* ws = (char*)d_ws;
    const size_t actBytes = (size_t)B_ * H_ * W_ * NCC_ * sizeof(_Float16); // 32 MB
    _Float16* actA  = (_Float16*)ws;
    _Float16* actB  = (_Float16*)(ws + actBytes);
    _Float16* wfrag = (_Float16*)(ws + 2 * actBytes);
    float*    biasH = (float*)(ws + 2 * actBytes + (size_t)WFRAG_BYTES);

    pack_hidden_weights<<<9, 256, 0, stream>>>(w_h, b_h, g_h, be_h, m_h, v_h,
                                               wfrag, biasH);
    input_conv_kernel<<<(B_ * H_ * W_ * NCC_) / 256, 256, 0, stream>>>(
        X, w_in, b_in, g_in, be_in, m_in, v_in, actA);

    _Float16* src = actA;
    _Float16* dst = actB;
    for (int l = 0; l < NHID; ++l) {
        hidden_conv_wmma<<<1024, 256, SMEM_BYTES, stream>>>(src, dst, wfrag,
                                                            biasH);
        _Float16* t = src; src = dst; dst = t;
    }

    output_conv_kernel<<<(B_ * H_ * W_) / 256, 256, 0, stream>>>(
        src, X, w_out, b_out, g_out, be_out, m_out, v_out, out);
}
